// MOCA_80075370266907
// MI455X (gfx1250) — matrix operand order repaired and checked by reading, not compiled, not run on hardware
//
#include <hip/hip_runtime.h>

typedef unsigned short u16;
typedef __attribute__((ext_vector_type(16))) __bf16 v16bf;
typedef __attribute__((ext_vector_type(8)))  float  v8f;

struct Pack32 { uint4 a, b; };
struct U16x4 { u16 x, y, z, w; };

// ---- CDNA5 async-to-LDS via inline asm (ASYNCcnt-tracked DMA into LDS)
__device__ __forceinline__ void async_cp16(const u16* g, const u16* l) {
  unsigned lds = (unsigned)(unsigned long long)l;   // flat aperture: addr[31:0] = LDS offset
  asm volatile("global_load_async_to_lds_b128 %0, %1, off"
               :: "v"(lds), "v"(g)
               : "memory");
}

#if defined(__has_builtin)
#if __has_builtin(__builtin_amdgcn_s_wait_asynccnt)
#define WAIT_ASYNC(n) __builtin_amdgcn_s_wait_asynccnt(n)
#endif
#endif
#ifndef WAIT_ASYNC
#define WAIT_ASYNC(n) asm volatile("s_wait_asynccnt %0" :: "i"(n) : "memory")
#endif

__device__ __forceinline__ u16 f2bf(float f) {
  unsigned u = __builtin_bit_cast(unsigned, f);
  u += 0x7fffu + ((u >> 16) & 1u);          // round-to-nearest-even
  return (u16)(u >> 16);
}
__device__ __forceinline__ float bf2f(u16 h) {
  unsigned u = ((unsigned)h) << 16;
  return __builtin_bit_cast(float, u);
}

__device__ __forceinline__ v16bf ld_frag(const u16* p0, const u16* p1) {
  Pack32 pk;
  pk.a = *(const uint4*)p0;
  pk.b = *(const uint4*)p1;
  return __builtin_bit_cast(v16bf, pk);
}

// ---------------- elementwise f32 -> bf16 ----------------
__global__ void f32_to_bf16_kernel(const float* __restrict__ in,
                                   u16* __restrict__ out, int n4) {
  int i = blockIdx.x * blockDim.x + threadIdx.x;
  if (i >= n4) return;
  float4 v = ((const float4*)in)[i];
  U16x4 o; o.x = f2bf(v.x); o.y = f2bf(v.y); o.z = f2bf(v.z); o.w = f2bf(v.w);
  ((U16x4*)out)[i] = o;
}

// ---------------- NT GEMM: C[M,N] = A[M,K] * Bt[N,K]^T ----------------
// 128(M) x 256(N) x 32(K) block, 256 threads = 8 wave32 (2x4), 64x64 wave tile,
// 4x4 = 16 v_wmma_f32_16x16x32_bf16 per wave per stage.
// Double-buffered LDS fed by GLOBAL_LOAD_ASYNC_TO_LDS_B128; peeled last stage.
template<int OUT_BF16, int BIAS_MODE /*0 none,1 over-N,2 over-M*/, int ADD_RES>
__global__ __launch_bounds__(256) void gemm_nt_bf16(
    const u16* __restrict__ A,  int lda, long long aB,
    const u16* __restrict__ Bt, int ldb, long long bB,
    void* __restrict__ Cv,      int ldc, long long cB,
    const float* __restrict__ bias,
    const float* __restrict__ res, long long rB,
    int K)
{
  constexpr int LSTR   = 40;             // 32 + 8 bf16 pad per row
  constexpr int BUFE_A = 128 * LSTR;     // elements per A tile buffer
  constexpr int BUFE_B = 256 * LSTR;     // elements per B tile buffer
  __shared__ __align__(16) u16 As[2 * BUFE_A];   // 20 KB
  __shared__ __align__(16) u16 Bs[2 * BUFE_B];   // 40 KB

  const int bz = blockIdx.z;
  const int bm = blockIdx.y * 128, bn = blockIdx.x * 256;
  const u16* Ag = A  + (long long)bz * aB + (long long)bm * lda;
  const u16* Bg = Bt + (long long)bz * bB + (long long)bn * ldb;

  const int tid  = threadIdx.x;
  const int lane = tid & 31;
  const int wv   = tid >> 5;
  const int wm   = (wv >> 2) * 64;   // 2 wave rows
  const int wn   = (wv & 3) * 64;    // 4 wave cols
  const int lr   = lane & 15;
  const int hi   = lane >> 4;

  // global<->LDS chunk mapping (8 bf16 per 16B chunk, 4 chunks per row)
  const int r0 = tid >> 2;
  const int cc = (tid & 3) << 3;

  v8f acc[4][4] = {};

  auto issue_stage = [&](int kt, int buf) {
    u16* Ad = As + buf * BUFE_A;
    u16* Bd = Bs + buf * BUFE_B;
    #pragma unroll
    for (int t = 0; t < 2; ++t) {         // A: 128 rows = 512 chunks
      int r = r0 + t * 64;
      async_cp16(&Ag[(long long)r * lda + kt + cc], &Ad[r * LSTR + cc]);
    }
    #pragma unroll
    for (int t = 0; t < 4; ++t) {         // B: 256 rows = 1024 chunks
      int r = r0 + t * 64;
      async_cp16(&Bg[(long long)r * ldb + kt + cc], &Bd[r * LSTR + cc]);
    }
  };

  auto compute_stage = [&](int buf) {
    const u16* Ab = As + buf * BUFE_A;
    const u16* Bb = Bs + buf * BUFE_B;
    v16bf af[4], bfr[4];
    #pragma unroll
    for (int m = 0; m < 4; ++m) {
      // A layout: lanes 0-15 hold K{0..7,16..23}, lanes 16-31 hold K{8..15,24..31}
      const u16* pa = &Ab[(wm + m * 16 + lr) * LSTR + hi * 8];
      af[m] = ld_frag(pa, pa + 16);
    }
    #pragma unroll
    for (int n = 0; n < 4; ++n) {
      // B layout: lanes 0-15 hold K{0..7,16..23}, lanes 16-31 hold K{8..15,24..31}
      const u16* pb = &Bb[(wn + n * 16 + lr) * LSTR + hi * 8];
      bfr[n] = ld_frag(pb, pb + 16);
    }
    #pragma unroll
    for (int m = 0; m < 4; ++m)
      #pragma unroll
      for (int n = 0; n < 4; ++n)
        acc[m][n] = __builtin_amdgcn_wmma_f32_16x16x32_bf16(
            false, af[m], false, bfr[n], (short)0, acc[m][n], false, false);
  };

  const int nst = K >> 5;
  issue_stage(0, 0);
  for (int s = 0; s < nst - 1; ++s) {     // steady state: branch-free pipeline
    issue_stage((s + 1) << 5, (s + 1) & 1);
    WAIT_ASYNC(6);                        // 6 newest = next stage; current done
    __syncthreads();
    compute_stage(s & 1);
    __syncthreads();
  }
  WAIT_ASYNC(0);
  __syncthreads();
  compute_stage((nst - 1) & 1);

  // epilogue: C VGPR j holds M = j (+8 for lanes>=16), N = lane&15
  #pragma unroll
  for (int m = 0; m < 4; ++m) {
    #pragma unroll
    for (int n = 0; n < 4; ++n) {
      int col = bn + wn + n * 16 + lr;
      float bN = (BIAS_MODE == 1) ? bias[col] : 0.0f;
      #pragma unroll
      for (int j = 0; j < 8; ++j) {
        int row = bm + wm + m * 16 + hi * 8 + j;
        float v = acc[m][n][j];
        if (BIAS_MODE == 1) v += bN;
        if (BIAS_MODE == 2) v += bias[row];
        if (ADD_RES) v += res[(long long)bz * rB + (long long)row * ldc + col];
        long long off = (long long)bz * cB + (long long)row * ldc + col;
        if (OUT_BF16) ((u16*)Cv)[off] = f2bf(v);
        else          ((float*)Cv)[off] = v;
      }
    }
  }
}

// ---------------- fused double-softmax (moca), transposed bf16 output ----------------
__device__ __forceinline__ float block_max(float v, float* red) {
  int tid = threadIdx.x;
  red[tid] = v; __syncthreads();
  #pragma unroll
  for (int s = 128; s > 0; s >>= 1) {
    if (tid < s) red[tid] = fmaxf(red[tid], red[tid + s]);
    __syncthreads();
  }
  float r = red[0]; __syncthreads();
  return r;
}
__device__ __forceinline__ float block_sum(float v, float* red) {
  int tid = threadIdx.x;
  red[tid] = v; __syncthreads();
  #pragma unroll
  for (int s = 128; s > 0; s >>= 1) {
    if (tid < s) red[tid] += red[tid + s];
    __syncthreads();
  }
  float r = red[0]; __syncthreads();
  return r;
}

__global__ __launch_bounds__(256) void moca_kernel(
    const u16* __restrict__ S1, const u16* __restrict__ S2,
    u16* __restrict__ mocaT,
    const float* __restrict__ rou_w, const float* __restrict__ rou_b)
{
  constexpr int T = 2048;
  __shared__ float red[256];
  const int t = blockIdx.x, b = blockIdx.y, tid = threadIdx.x;

  // faithful torch cat/view interleave: moca[0]=f(nssm0,nssm1), moca[1]=f(nssm2,nssm3),
  //                                     moca[2]=f(att0,att1),   moca[3]=f(att2,att3)
  const u16* src = (b < 2) ? S1 : S2;
  const int  b0  = (b < 2) ? (2 * b) : (2 * (b - 2));
  const u16* rowA = src + ((long long)b0 * T + t) * T;
  const u16* rowB = src + ((long long)(b0 + 1) * T + t) * T;

  float a[8], c[8];
  #pragma unroll
  for (int i = 0; i < 8; ++i) {
    int s = tid + i * 256;
    a[i] = bf2f(rowA[s]);
    c[i] = bf2f(rowB[s]);
  }
  float mA = -1e30f, mB = -1e30f;
  #pragma unroll
  for (int i = 0; i < 8; ++i) { mA = fmaxf(mA, a[i]); mB = fmaxf(mB, c[i]); }
  mA = block_max(mA, red);
  mB = block_max(mB, red);

  float sA = 0.f, sB = 0.f;
  #pragma unroll
  for (int i = 0; i < 8; ++i) {
    a[i] = __expf(a[i] - mA); sA += a[i];
    c[i] = __expf(c[i] - mB); sB += c[i];
  }
  sA = block_sum(sA, red);
  sB = block_sum(sB, red);

  const float r0 = rou_w[0], r1 = rou_w[1], rb = rou_b[0];
  const float iA = 1.f / sA, iB = 1.f / sB;

  float p[8], mp = -1e30f;
  #pragma unroll
  for (int i = 0; i < 8; ++i) {
    p[i] = r0 * a[i] * iA + r1 * c[i] * iB + rb;
    mp = fmaxf(mp, p[i]);
  }
  mp = block_max(mp, red);

  float sp = 0.f;
  #pragma unroll
  for (int i = 0; i < 8; ++i) { p[i] = __expf(p[i] - mp); sp += p[i]; }
  sp = block_sum(sp, red);
  const float inv = 1.f / sp;

  #pragma unroll
  for (int i = 0; i < 8; ++i) {
    int s = tid + i * 256;
    mocaT[((long long)b * T + s) * T + t] = f2bf(p[i] * inv);  // transposed store
  }
}

// ---------------- launch ----------------
extern "C" void kernel_launch(void* const* d_in, const int* in_sizes, int n_in,
                              void* d_out, int out_size, void* d_ws, size_t ws_size,
                              hipStream_t stream) {
  (void)in_sizes; (void)n_in; (void)out_size; (void)ws_size;
  constexpr int B = 4, T = 2048, CIN = 2048, CHID = 1024, CEMB = 256;

  const float* x       = (const float*)d_in[0];
  const float* theta_w = (const float*)d_in[1];
  const float* theta_b = (const float*)d_in[2];
  const float* phi_w   = (const float*)d_in[3];
  const float* phi_b   = (const float*)d_in[4];
  const float* g_w     = (const float*)d_in[5];
  const float* g_b     = (const float*)d_in[6];
  const float* rou_w   = (const float*)d_in[7];
  const float* rou_b   = (const float*)d_in[8];
  const float* w_w     = (const float*)d_in[9];
  const float* w_b     = (const float*)d_in[10];
  const float* emb_w   = (const float*)d_in[11];
  const float* emb_b   = (const float*)d_in[12];
  float* out = (float*)d_out;

  char* p = (char*)d_ws;
  auto alloc = [&](size_t bytes) -> char* {
    char* r = p; p += (bytes + 255) & ~(size_t)255; return r;
  };
  u16* Xbf    = (u16*)alloc((size_t)B * T * CIN * 2);
  u16* thW    = (u16*)alloc((size_t)CHID * CIN * 2);
  u16* phW    = (u16*)alloc((size_t)CHID * CIN * 2);
  u16* gW     = (u16*)alloc((size_t)CHID * CIN * 2);
  u16* wW     = (u16*)alloc((size_t)CIN * CHID * 2);
  u16* embW   = (u16*)alloc((size_t)CEMB * CIN * 2);
  u16* thetaT = (u16*)alloc((size_t)B * T * CHID * 2);   // (T,H) per batch
  u16* phiT   = (u16*)alloc((size_t)B * T * CHID * 2);   // (T,H)
  u16* xg     = (u16*)alloc((size_t)B * CHID * T * 2);   // (H,T)
  u16* S1     = (u16*)alloc((size_t)B * T * T * 2);      // raw self-sim scores
  u16* S2     = (u16*)alloc((size_t)B * T * T * 2);      // raw att scores
  u16* mocaT  = (u16*)alloc((size_t)B * T * T * 2);      // (s,t) transposed
  u16* yT     = (u16*)alloc((size_t)B * T * CHID * 2);   // (T,H)
  u16* zbf    = (u16*)alloc((size_t)B * T * CIN * 2);    // (T,C)

  auto cvt = [&](const float* src, u16* dst, size_t n) {
    int n4 = (int)(n / 4);
    f32_to_bf16_kernel<<<(n4 + 255) / 256, 256, 0, stream>>>(src, dst, n4);
  };
  cvt(x,       Xbf,  (size_t)B * T * CIN);
  cvt(theta_w, thW,  (size_t)CHID * CIN);
  cvt(phi_w,   phW,  (size_t)CHID * CIN);
  cvt(g_w,     gW,   (size_t)CHID * CIN);
  cvt(w_w,     wW,   (size_t)CIN * CHID);
  cvt(emb_w,   embW, (size_t)CEMB * CIN);

  dim3 blk(256);
  const long long sXC = (long long)T * CIN;
  const long long sTH = (long long)T * CHID;
  const long long sHT = (long long)CHID * T;
  const long long sTT = (long long)T * T;

  // thetaT[t,h] = sum_c X[t,c]*theta_w[h,c] + theta_b[h]
  gemm_nt_bf16<1, 1, 0><<<dim3(CHID / 256, T / 128, B), blk, 0, stream>>>(
      Xbf, CIN, sXC, thW, CIN, 0, thetaT, CHID, sTH, theta_b, nullptr, 0, CIN);
  // phiT[t,h]
  gemm_nt_bf16<1, 1, 0><<<dim3(CHID / 256, T / 128, B), blk, 0, stream>>>(
      Xbf, CIN, sXC, phW, CIN, 0, phiT, CHID, sTH, phi_b, nullptr, 0, CIN);
  // xg[h,t] = sum_c g_w[h,c]*X[t,c] + g_b[h]   (bias over M)
  gemm_nt_bf16<1, 2, 0><<<dim3(T / 256, CHID / 128, B), blk, 0, stream>>>(
      gW, CIN, 0, Xbf, CIN, sXC, xg, T, sHT, g_b, nullptr, 0, CIN);
  // S1[t,s] = X[t,:] . X[s,:]
  gemm_nt_bf16<1, 0, 0><<<dim3(T / 256, T / 128, B), blk, 0, stream>>>(
      Xbf, CIN, sXC, Xbf, CIN, sXC, S1, T, sTT, nullptr, nullptr, 0, CIN);
  // S2[t,s] = phiT[t,:] . thetaT[s,:]
  gemm_nt_bf16<1, 0, 0><<<dim3(T / 256, T / 128, B), blk, 0, stream>>>(
      phiT, CHID, sTH, thetaT, CHID, sTH, S2, T, sTT, nullptr, nullptr, 0, CHID);
  // moca (double softmax with batch interleave), transposed output
  moca_kernel<<<dim3(T, B), blk, 0, stream>>>(S1, S2, mocaT, rou_w, rou_b);
  // yT[s,h] = sum_t mocaT[s,t]*xg[h,t]
  gemm_nt_bf16<1, 0, 0><<<dim3(CHID / 256, T / 128, B), blk, 0, stream>>>(
      mocaT, T, sTT, xg, T, sHT, yT, CHID, sTH, nullptr, nullptr, 0, T);
  // z[t,o] = sum_h yT[t,h]*w_w[o,h] + w_b[o] + x[t,o]
  gemm_nt_bf16<1, 1, 1><<<dim3(CIN / 256, T / 128, B), blk, 0, stream>>>(
      yT, CHID, sTH, wW, CHID, 0, zbf, CIN, sXC, w_b, x, sXC, CHID);
  // out[t,e] = sum_c z[t,c]*emb_w[e,c] + emb_b[e]   (fp32 output)
  gemm_nt_bf16<0, 1, 0><<<dim3(CEMB / 256, T / 128, B), blk, 0, stream>>>(
      zbf, CIN, sXC, embW, CIN, 0, out, CEMB, (long long)T * CEMB, emb_b, nullptr, 0, CIN);
}
